// DEA_CARAFE_8443905704234
// MI455X (gfx1250) — matrix operand order repaired and checked by reading, not compiled, not run on hardware
//
#include <hip/hip_runtime.h>
#include <math.h>

// ---------- CDNA5 WMMA types ----------
typedef __attribute__((ext_vector_type(16))) __bf16 v16bf;
typedef __attribute__((ext_vector_type(8)))  float  v8f;

#define C_IN   128
#define CMID   64
#define H      64
#define Wd     64
#define NPIX   (2*H*Wd)           // 8192
#define MENC   128                // 100 padded to 128
#define HP     68                 // 64 + 2*2 pad (5x5 conv)
#define WP     68
#define HP1    66                 // 64 + 2*1 pad (3x3 conv)
#define WP1    66

#define WMMA_BF16(A, B, C) \
  __builtin_amdgcn_wmma_f32_16x16x32_bf16(false, (A), false, (B), (short)0, (C), false, false)

// Fragment permutation for 16-bit WMMA A-matrix: within each 32-K block,
// store K in order [0..7, 16..23, 8..15, 24..31] so that
//   lane<16  reads bytes [0..31]  -> K {0..7,16..23}   (elements 0..15)
//   lane>=16 reads bytes [32..63] -> K {8..15,24..31}
// i.e. one contiguous 32B load per lane, matching the ISA VGPR layout.
__device__ __host__ inline int fragperm(int p) {
  return (p < 8) ? p : (p < 16) ? (p + 8) : (p < 24) ? (p - 8) : p;
}

// ============================================================
// Kernel 1: weight prep (both GEMM A-panels, K in natural order).
//  Aw [25][64][128] bf16 : tap-major combined (w5 + embedded w3)
//  Ae [ 9][128][64] bf16 : encoder weights, rows 100..127 zero
// ============================================================
__global__ void prep_weights(const float* __restrict__ w3,
                             const float* __restrict__ w5,
                             const float* __restrict__ wenc,
                             __bf16* __restrict__ Aw,
                             __bf16* __restrict__ Ae) {
  int idx = blockIdx.x * blockDim.x + threadIdx.x;
  if (idx < 25 * CMID * C_IN) {                 // 204800
    int t = idx >> 13, rem = idx & 8191;        // /8192
    int m = rem >> 7, p = rem & 127;
    int ci = p;
    int ky = t / 5, kx = t - ky * 5;
    float v = w5[((m * C_IN + ci) * 5 + ky) * 5 + kx];
    if (ky >= 1 && ky <= 3 && kx >= 1 && kx <= 3)
      v += w3[((m * C_IN + ci) * 3 + (ky - 1)) * 3 + (kx - 1)];
    Aw[idx] = (__bf16)v;
  }
  if (idx < 9 * MENC * CMID) {                  // 73728
    int t = idx >> 13, rem = idx & 8191;
    int m = rem >> 6, p = rem & 63;
    int c = p;
    float v = (m < 100) ? wenc[(m * CMID + c) * 9 + t] : 0.0f;
    Ae[idx] = (__bf16)v;
  }
}

// ============================================================
// Kernel 2: X -> NHWC bf16 with 2-pixel zero pad: (2,68,68,128)
// ============================================================
__global__ void prep_x_nhwc(const float* __restrict__ X, __bf16* __restrict__ Xn) {
  int idx = blockIdx.x * blockDim.x + threadIdx.x;
  const int tot = 2 * HP * WP * C_IN;
  if (idx >= tot) return;
  int b = idx / (HP * WP * C_IN), rem = idx - b * (HP * WP * C_IN);
  int pos = rem >> 7, ci = rem & 127;
  int y = pos / WP, x = pos - y * WP;
  float v = 0.0f;
  if (y >= 2 && y < HP - 2 && x >= 2 && x < WP - 2)
    v = X[(((size_t)b * C_IN + ci) * H + (y - 2)) * Wd + (x - 2)];
  Xn[idx] = (__bf16)v;
}

// ============================================================
// Kernel 3: fused (3x3+5x5) conv as direct-conv WMMA, no LDS.
//  Per tap: GEMM K=128 channels (4 x 32-K WMMA steps).
//  Block = one image row (64 px), 128 thr = 4 waves.
//  Each wave owns a 32M x 32N tile (2x2 WMMA tiles): both A and B
//  fragments feed two WMMAs -> 8 b128 loads per 4 WMMAs.
// ============================================================
__global__ __launch_bounds__(128) void conv5_wmma(
    const __bf16* __restrict__ Aw,    // [25][64][128] natural K order
    const __bf16* __restrict__ Xn,    // [2][68][68][128] NHWC
    const float* __restrict__ b3, const float* __restrict__ b5,
    float* __restrict__ Wout)         // [2][64][64][64] NCHW f32
{
  const int tid = threadIdx.x, lane = tid & 31, wave = tid >> 5;  // 4 waves
  const int n0 = blockIdx.x * 64;
  const int bb = n0 >> 12, py = (n0 & 4095) >> 6;   // batch, row
  const int mpair = wave >> 1, npair = wave & 1;
  const int lx = lane & 15;
  const int hoff = (lane < 16) ? 0 : 16;            // fragment half offset (elems)
  const int x0 = npair * 32 + lx, x1 = x0 + 16;     // N columns
  const int m0 = mpair * 32 + lx;                   // A rows (m1 = m0+16)

  v8f acc00 = {}, acc01 = {}, acc10 = {}, acc11 = {};
  const __bf16* arow0 = Aw + (size_t)m0 * C_IN;
  const __bf16* arow1 = arow0 + 16 * C_IN;
  const __bf16* xrow  = Xn + ((size_t)bb * HP + py) * WP * C_IN;

  for (int t = 0; t < 25; ++t) {
    const int ky = t / 5, kx = t - ky * 5;
    const size_t tA = (size_t)t * (CMID * C_IN);
    const __bf16* a0  = arow0 + tA;
    const __bf16* a1  = arow1 + tA;
    const __bf16* b0p = xrow + ((size_t)ky * WP + x0 + kx) * C_IN;
    const __bf16* b1p = xrow + ((size_t)ky * WP + x1 + kx) * C_IN;
    if (t + 1 < 25)
      __builtin_prefetch(arow0 + (size_t)(t + 1) * (CMID * C_IN), 0, 1);
    #pragma unroll
    for (int ks = 0; ks < 4; ++ks) {
      v16bf af0 = *(const v16bf*)(a0  + ks * 32 + hoff);
      v16bf af1 = *(const v16bf*)(a1  + ks * 32 + hoff);
      v16bf bf0 = *(const v16bf*)(b0p + ks * 32 + hoff);
      v16bf bf1 = *(const v16bf*)(b1p + ks * 32 + hoff);
      acc00 = WMMA_BF16(af0, bf0, acc00);
      acc01 = WMMA_BF16(af0, bf1, acc01);
      acc10 = WMMA_BF16(af1, bf0, acc10);
      acc11 = WMMA_BF16(af1, bf1, acc11);
    }
  }

  // D layout: VGPR r -> M = r (+8 for lanes 16..31), lane&15 = N column
  #pragma unroll
  for (int r = 0; r < 8; ++r) {
    const int mo0 = mpair * 32 + r + ((lane >= 16) ? 8 : 0);
    const int mo1 = mo0 + 16;
    const float bias0 = b3[mo0] + b5[mo0];
    const float bias1 = b3[mo1] + b5[mo1];
    float* wr0 = Wout + (((size_t)bb * CMID + mo0) * H + py) * Wd;
    float* wr1 = Wout + (((size_t)bb * CMID + mo1) * H + py) * Wd;
    wr0[x0] = acc00[r] + bias0;
    wr0[x1] = acc01[r] + bias0;
    wr1[x0] = acc10[r] + bias1;
    wr1[x1] = acc11[r] + bias1;
  }
}

// ============================================================
// Kernel 4: SE gate.  grid=2 (batch), 256 threads.
// ============================================================
__global__ __launch_bounds__(256) void se_gate(
    const float* __restrict__ Wb,
    const float* __restrict__ wa1, const float* __restrict__ ba1,
    const float* __restrict__ wa2, const float* __restrict__ ba2,
    float* __restrict__ gate)
{
  __shared__ float partial[256];
  __shared__ float avg[CMID];
  __shared__ float hid[8];
  const int b = blockIdx.x, t = threadIdx.x;
  const int c = t & 63, part = t >> 6;
  const float* wc = Wb + ((size_t)b * CMID + c) * 4096;
  float s = 0.f;
  for (int i = part * 1024; i < part * 1024 + 1024; ++i) s += wc[i];
  partial[t] = s;
  __syncthreads();
  if (t < CMID)
    avg[t] = (partial[t] + partial[t + 64] + partial[t + 128] + partial[t + 192]) * (1.0f / 4096.0f);
  __syncthreads();
  if (t < 8) {
    float h = ba1[t];
    for (int c2 = 0; c2 < CMID; ++c2) h += wa1[t * CMID + c2] * avg[c2];
    hid[t] = h;
  }
  __syncthreads();
  if (t < CMID) {
    float g = ba2[t];
    for (int j = 0; j < 8; ++j) g += wa2[t * 8 + j] * hid[j];
    gate[b * CMID + t] = 1.0f / (1.0f + __expf(-g));
  }
}

// ============================================================
// Kernel 5: W*gate + depthwise 3x3 edge -> NHWC bf16 with 1-pad
//  (zero border): Wn (2,66,66,64). grid = 128 (b*64+c).
// ============================================================
__global__ __launch_bounds__(256) void attn_edge(
    const float* __restrict__ Wb, const float* __restrict__ gate,
    const float* __restrict__ wedge, __bf16* __restrict__ Wn)
{
  const int bc = blockIdx.x;
  const int b = bc >> 6, c = bc & 63;
  const float* src = Wb + (size_t)bc * 4096;
  const float g = gate[bc];
  float ker[9];
  #pragma unroll
  for (int i = 0; i < 9; ++i) ker[i] = wedge[c * 9 + i];
  __bf16* dst = Wn + (size_t)b * (HP1 * WP1 * CMID) + c;

  for (int p = threadIdx.x; p < HP1 * WP1; p += 256) {
    const int y = p / WP1, x = p - y * WP1;
    float out = 0.f;
    if (y >= 1 && y < HP1 - 1 && x >= 1 && x < HP1 - 1) {
      const int yy = y - 1, xx = x - 1;
      float e = 0.f;
      #pragma unroll
      for (int ky = 0; ky < 3; ++ky)
        #pragma unroll
        for (int kx = 0; kx < 3; ++kx) {
          const int sy = yy + ky - 1, sx = xx + kx - 1;
          if (sy >= 0 && sy < H && sx >= 0 && sx < Wd)
            e += ker[ky * 3 + kx] * src[sy * Wd + sx];
        }
      out = src[yy * Wd + xx] * g + e;
    }
    dst[(size_t)p * CMID] = (__bf16)out;
  }
}

// ============================================================
// Kernel 6: encoder 3x3 conv as direct-conv WMMA, no LDS.
//  M=128 (100 real), K=64 per tap (2 x 32-K steps), 9 taps.
//  Block = 32 px, 128 thr = 4 waves; wave owns 32M x 32N (2x2 tiles).
// ============================================================
__global__ __launch_bounds__(128) void enc_wmma(
    const __bf16* __restrict__ Ae,    // [9][128][64] natural K order
    const __bf16* __restrict__ Wn,    // [2][66][66][64] NHWC
    float* __restrict__ Enc)          // [2][100][64][64] NCHW f32
{
  const int tid = threadIdx.x, lane = tid & 31, wave = tid >> 5;  // 4 waves
  const int n0 = blockIdx.x * 32;
  const int bb = n0 >> 12, py = (n0 & 4095) >> 6;
  const int xb = n0 & 63;                           // 0 or 32
  const int lx = lane & 15;
  const int hoff = (lane < 16) ? 0 : 16;
  const int x0 = xb + lx, x1 = x0 + 16;
  const int m0 = wave * 32 + lx;                    // m1 = m0+16

  v8f acc00 = {}, acc01 = {}, acc10 = {}, acc11 = {};
  const __bf16* arow0 = Ae + (size_t)m0 * CMID;
  const __bf16* arow1 = arow0 + 16 * CMID;
  const __bf16* wrow  = Wn + ((size_t)bb * HP1 + py) * WP1 * CMID;

  for (int t = 0; t < 9; ++t) {
    const int ky = t / 3, kx = t - ky * 3;
    const size_t tA = (size_t)t * (MENC * CMID);
    const __bf16* a0  = arow0 + tA;
    const __bf16* a1  = arow1 + tA;
    const __bf16* b0p = wrow + ((size_t)ky * WP1 + x0 + kx) * CMID;
    const __bf16* b1p = wrow + ((size_t)ky * WP1 + x1 + kx) * CMID;
    #pragma unroll
    for (int ks = 0; ks < 2; ++ks) {
      v16bf af0 = *(const v16bf*)(a0  + ks * 32 + hoff);
      v16bf af1 = *(const v16bf*)(a1  + ks * 32 + hoff);
      v16bf bf0 = *(const v16bf*)(b0p + ks * 32 + hoff);
      v16bf bf1 = *(const v16bf*)(b1p + ks * 32 + hoff);
      acc00 = WMMA_BF16(af0, bf0, acc00);
      acc01 = WMMA_BF16(af0, bf1, acc01);
      acc10 = WMMA_BF16(af1, bf0, acc10);
      acc11 = WMMA_BF16(af1, bf1, acc11);
    }
  }

  #pragma unroll
  for (int r = 0; r < 8; ++r) {
    const int mo0 = wave * 32 + r + ((lane >= 16) ? 8 : 0);
    const int mo1 = mo0 + 16;
    if (mo0 < 100) {
      float* er = Enc + (((size_t)bb * 100 + mo0) * H + py) * Wd;
      er[x0] = acc00[r];
      er[x1] = acc01[r];
    }
    if (mo1 < 100) {
      float* er = Enc + (((size_t)bb * 100 + mo1) * H + py) * Wd;
      er[x0] = acc10[r];
      er[x1] = acc11[r];
    }
  }
}

// ============================================================
// Kernel 7: fused pixelshuffle + softmax(25) + CARAFE.
//  Stages the 100 logits into LDS via CDNA5 async global->LDS DMA.
//  out[b,c,2y+sy,2x+sx] = sum_{i,j} p[sy,sx][i*5+j]*X[b,c,y+i-2,x+j-2]
// ============================================================
__global__ __launch_bounds__(128) void carafe_fused(
    const float* __restrict__ Enc, const float* __restrict__ X,
    float* __restrict__ out)
{
  __shared__ float p[4][25];
  const int blk = blockIdx.x;
  const int b = blk >> 12, rem = blk & 4095;
  const int y = rem >> 6, x = rem & 63;
  const int t = threadIdx.x;

  if (t < 100) {
    const int s = t & 3, tap = t >> 2;      // enc channel = tap*4 + sy*2 + sx
    const float* g = Enc + (((size_t)b * 100 + (tap * 4 + s)) * H + y) * Wd + x;
    // CDNA5 async global->LDS copy (per-lane, tracked by ASYNCcnt).
    // Generic LDS pointer low 32 bits == LDS byte offset (ISA 10.2).
    unsigned loff = (unsigned)(uintptr_t)&p[s][tap];
    unsigned long long gaddr = (unsigned long long)(uintptr_t)g;
    asm volatile("global_load_async_to_lds_b32 %0, %1, off"
                 :: "v"(loff), "v"(gaddr) : "memory");
  }
  asm volatile("s_wait_asynccnt 0" ::: "memory");
  __syncthreads();

  if (t < 4) {
    float mx = p[t][0];
    #pragma unroll
    for (int i = 1; i < 25; ++i) mx = fmaxf(mx, p[t][i]);
    float sum = 0.f;
    #pragma unroll
    for (int i = 0; i < 25; ++i) { float e = __expf(p[t][i] - mx); p[t][i] = e; sum += e; }
    const float inv = 1.0f / sum;
    #pragma unroll
    for (int i = 0; i < 25; ++i) p[t][i] *= inv;
  }
  __syncthreads();

  const float* xc = X + ((size_t)b * C_IN + t) * 4096;
  float xv[25];
  #pragma unroll
  for (int i = 0; i < 5; ++i)
    #pragma unroll
    for (int j = 0; j < 5; ++j) {
      const int sy = y + i - 2, sx = x + j - 2;
      xv[i * 5 + j] = (sy >= 0 && sy < 64 && sx >= 0 && sx < 64) ? xc[sy * 64 + sx] : 0.f;
    }
  #pragma unroll
  for (int s = 0; s < 4; ++s) {
    float acc = 0.f;
    #pragma unroll
    for (int k = 0; k < 25; ++k) acc += p[s][k] * xv[k];
    out[(((size_t)b * C_IN + t) * 128 + (2 * y + (s >> 1))) * 128 + (2 * x + (s & 1))] = acc;
  }
}

// ============================================================
extern "C" void kernel_launch(void* const* d_in, const int* in_sizes, int n_in,
                              void* d_out, int out_size, void* d_ws, size_t ws_size,
                              hipStream_t stream) {
  (void)in_sizes; (void)n_in; (void)out_size; (void)ws_size;
  const float* X     = (const float*)d_in[0];
  const float* w3    = (const float*)d_in[1];
  const float* b3    = (const float*)d_in[2];
  const float* w5    = (const float*)d_in[3];
  const float* b5    = (const float*)d_in[4];
  const float* wa1   = (const float*)d_in[5];
  const float* ba1   = (const float*)d_in[6];
  const float* wa2   = (const float*)d_in[7];
  const float* ba2   = (const float*)d_in[8];
  const float* wedge = (const float*)d_in[9];
  const float* wenc  = (const float*)d_in[10];
  float* out = (float*)d_out;

  char* ws = (char*)d_ws;
  size_t off = 0;
  auto alloc = [&](size_t bytes) {
    void* ptr = ws + off;
    off += (bytes + 255) & ~(size_t)255;
    return ptr;
  };
  __bf16* Xn   = (__bf16*)alloc((size_t)2 * HP * WP * C_IN * 2);    // 2.37 MB NHWC
  __bf16* Aw   = (__bf16*)alloc((size_t)25 * CMID * C_IN * 2);      // 0.41 MB
  __bf16* Ae   = (__bf16*)alloc((size_t)9 * MENC * CMID * 2);       // 0.15 MB
  float*  Wbuf = (float*)alloc((size_t)2 * CMID * H * Wd * 4);      // 2.10 MB
  float*  gate = (float*)alloc((size_t)2 * CMID * 4);
  __bf16* Wn   = (__bf16*)alloc((size_t)2 * HP1 * WP1 * CMID * 2);  // 1.11 MB NHWC
  float*  Enc  = (float*)alloc((size_t)2 * 100 * H * Wd * 4);       // 3.28 MB

  prep_weights<<<(25 * CMID * C_IN + 255) / 256, 256, 0, stream>>>(w3, w5, wenc, Aw, Ae);
  prep_x_nhwc<<<(2 * HP * WP * C_IN + 255) / 256, 256, 0, stream>>>(X, Xn);
  conv5_wmma<<<NPIX / 64, 128, 0, stream>>>(Aw, Xn, b3, b5, Wbuf);
  se_gate<<<2, 256, 0, stream>>>(Wbuf, wa1, ba1, wa2, ba2, gate);
  attn_edge<<<2 * CMID, 256, 0, stream>>>(Wbuf, gate, wedge, Wn);
  enc_wmma<<<NPIX / 32, 128, 0, stream>>>(Ae, Wn, Enc);
  carafe_fused<<<2 * H * Wd, 128, 0, stream>>>(Enc, X, out);
}
